// Generator_9887014715849
// MI455X (gfx1250) — matrix operand order repaired and checked by reading, not compiled, not run on hardware
//
#include <hip/hip_runtime.h>

// ---------------------------------------------------------------------------
// MI455X (gfx1250) implementation.
// Dominant work: 3x (10000x10000 adj) @ (10000x{256,256,128}) = 128 GFLOP.
// Strategy: bf16 operands + V_WMMA_F32_16X16X32_BF16, f32 accumulate.
// adj streamed from HBM as f32, converted to bf16 while staging to LDS
// (3 x 400MB reads ~= 51us floor at 23.3 TB/s; bf16 WMMA rate >> that).
// Kernels are templated so the hot K-loop carries no runtime-flag branches;
// only the single M-edge block row / K-tail chunk take the guarded path.
// ---------------------------------------------------------------------------

typedef __bf16 v16bf __attribute__((ext_vector_type(16)));
typedef __bf16 v4bf  __attribute__((ext_vector_type(4)));
typedef float  v8f   __attribute__((ext_vector_type(8)));

#define BM 128
#define BN 128
#define BK 32

// ---------------------------------------------------------------------------
// WMMA GEMM: C = A @ B (+bias, +relu).
//   A: MxK row-major, f32 (AF32) or bf16, leading dim lda
//   B: KxN row-major bf16, leading dim ldb   (requires N % 128 == 0)
//   C: MxN, leading dim ldc, f32 or bf16 (CBF16)
// 128x128 block tile, K-step 32; 8 waves, each owns 32x64 = 2x4 WMMA tiles.
// ---------------------------------------------------------------------------
template <bool AF32, bool CBF16, bool RELU>
__global__ __launch_bounds__(256)
void gemm_wmma_kernel(const void* __restrict__ A_,
                      const __bf16* __restrict__ B,
                      const float* __restrict__ bias,
                      void* __restrict__ C_,
                      int M, int N, int K, int lda, int ldb, int ldc)
{
    __shared__ __align__(16) __bf16 As[BM * BK];    // row-major 128x32
    __shared__ __align__(16) __bf16 Bst[BN * BK];   // transposed: [col][k]

    const int tid  = threadIdx.x;
    const int lane = tid & 31;
    const int wave = tid >> 5;
    const int wm   = wave >> 1;        // 0..3 : 32-row strip
    const int wn   = wave & 1;         // 0..1 : 64-col strip
    const int m16  = lane & 15;
    const int hh   = lane >> 4;        // half-wave select

    const int rowBase = blockIdx.y * BM;
    const int colBase = blockIdx.x * BN;
    const bool interiorM = (rowBase + BM <= M);

    v8f acc[2][4];
#pragma unroll
    for (int i = 0; i < 2; ++i)
#pragma unroll
        for (int j = 0; j < 4; ++j)
            acc[i][j] = (v8f){0.f,0.f,0.f,0.f,0.f,0.f,0.f,0.f};

    const float*  Af = (const float*) A_;
    const __bf16* Ab = (const __bf16*)A_;

    // B-fill mapping: thread covers 2 consecutive k-rows x 8 cols
    const int bC0 = (tid & 15) * 8;    // 0..120
    const int bK0 = (tid >> 4) * 2;    // 0,2,..,30

    for (int k0 = 0; k0 < K; k0 += BK) {
        const bool fullK = (k0 + BK <= K);

        if (interiorM && fullK) {
            // ---------------- fast staging: no guards ----------------
            if (AF32) {
#pragma unroll
                for (int p = 0; p < 4; ++p) {
                    int li = p * 1024 + tid * 4;
                    int r = li >> 5, c = li & 31;
                    const float* ap = Af + (size_t)(rowBase + r) * lda + (k0 + c);
                    float4 v = *(const float4*)ap;
                    if (c == 0 && k0 + BK < K)      // one prefetch per row-line
                        __builtin_prefetch(ap + BK, 0, 3);
                    v4bf pk = { (__bf16)v.x, (__bf16)v.y, (__bf16)v.z, (__bf16)v.w };
                    *(v4bf*)&As[r * BK + c] = pk;
                }
            } else {
#pragma unroll
                for (int p = 0; p < 2; ++p) {
                    int li = p * 2048 + tid * 8;
                    int r = li >> 5, c = li & 31;
                    *(uint4*)&As[r * BK + c] =
                        *(const uint4*)(Ab + (size_t)(rowBase + r) * lda + (k0 + c));
                }
            }
            {
                const __bf16* bp = B + (size_t)(k0 + bK0) * ldb + colBase + bC0;
                uint4 q0 = *(const uint4*)bp;
                uint4 q1 = *(const uint4*)(bp + ldb);
                const __bf16* b0 = (const __bf16*)&q0;
                const __bf16* b1 = (const __bf16*)&q1;
#pragma unroll
                for (int j = 0; j < 8; ++j) {
                    union { __bf16 h[2]; unsigned u; } pk;
                    pk.h[0] = b0[j]; pk.h[1] = b1[j];
                    *(unsigned*)&Bst[(bC0 + j) * BK + bK0] = pk.u;
                }
            }
        } else {
            // ---------------- guarded staging (edge blocks only) ----------------
            if (AF32) {
#pragma unroll
                for (int p = 0; p < 4; ++p) {
                    int li = p * 1024 + tid * 4;
                    int r = li >> 5, c = li & 31;
                    int gr = rowBase + r, gc = k0 + c;
#pragma unroll
                    for (int j = 0; j < 4; ++j) {
                        float v = (gr < M && gc + j < K)
                                ? Af[(size_t)gr * lda + gc + j] : 0.f;
                        As[r * BK + c + j] = (__bf16)v;
                    }
                }
            } else {
#pragma unroll
                for (int p = 0; p < 2; ++p) {
                    int li = p * 2048 + tid * 8;
                    int r = li >> 5, c = li & 31;
                    int gr = rowBase + r, gc = k0 + c;
#pragma unroll
                    for (int j = 0; j < 8; ++j)
                        As[r * BK + c + j] = (gr < M && gc + j < K)
                            ? Ab[(size_t)gr * lda + gc + j] : (__bf16)0.0f;
                }
            }
            {
#pragma unroll
                for (int kk = 0; kk < 2; ++kk) {
                    int gk = k0 + bK0 + kk;
#pragma unroll
                    for (int j = 0; j < 8; ++j)
                        Bst[(bC0 + j) * BK + bK0 + kk] = (gk < K)
                            ? B[(size_t)gk * ldb + colBase + bC0 + j] : (__bf16)0.0f;
                }
            }
        }
        __syncthreads();

        // ---- fragment loads (ds_load_b128) + 8 WMMAs per wave ----
        union { v16bf v; uint4 q[2]; } afr[2], bfr[4];
#pragma unroll
        for (int mi = 0; mi < 2; ++mi) {
            int r = wm * 32 + mi * 16 + m16;
            // A 16-bit layout: half hh: K = hh*8 + j (j<8), 16 + hh*8 + (j-8)
            afr[mi].q[0] = *(const uint4*)&As[r * BK + hh * 8];
            afr[mi].q[1] = *(const uint4*)&As[r * BK + 16 + hh * 8];
        }
#pragma unroll
        for (int ni = 0; ni < 4; ++ni) {
            int cc = wn * 64 + ni * 16 + m16;
            // B layout: half hh: K = hh*8 + j (j<8), 16 + hh*8 + (j-8)
            bfr[ni].q[0] = *(const uint4*)&Bst[cc * BK + hh * 8];
            bfr[ni].q[1] = *(const uint4*)&Bst[cc * BK + 16 + hh * 8];
        }
#pragma unroll
        for (int mi = 0; mi < 2; ++mi)
#pragma unroll
            for (int ni = 0; ni < 4; ++ni)
                acc[mi][ni] = __builtin_amdgcn_wmma_f32_16x16x32_bf16(
                    false, afr[mi].v, false, bfr[ni].v,
                    (short)0, acc[mi][ni], false, false);
        __syncthreads();
    }

    // ---- epilogue: C layout VGPR r -> (M = hh*8 + r, N = m16) ----
#pragma unroll
    for (int mi = 0; mi < 2; ++mi) {
#pragma unroll
        for (int ni = 0; ni < 4; ++ni) {
            int gcol = colBase + wn * 64 + ni * 16 + m16;
            float bv = bias ? bias[gcol] : 0.0f;
#pragma unroll
            for (int r8 = 0; r8 < 8; ++r8) {
                int grow = rowBase + wm * 32 + mi * 16 + hh * 8 + r8;
                if (grow >= M) continue;
                float v = acc[mi][ni][r8] + bv;
                if (RELU && v < 0.f) v = 0.f;
                if (CBF16) ((__bf16*)C_)[(size_t)grow * ldc + gcol] = (__bf16)v;
                else       ((float*)C_) [(size_t)grow * ldc + gcol] = v;
            }
        }
    }
}

// f32 sub-matrix -> bf16 (weights, x slices, concat tail)
__global__ void cvt_submat_kernel(const float* __restrict__ src, int src_ld, int col0,
                                  __bf16* __restrict__ dst, int dst_ld,
                                  int rows, int cols)
{
    int idx = blockIdx.x * blockDim.x + threadIdx.x;
    if (idx < rows * cols) {
        int r = idx / cols, c = idx - r * cols;
        dst[(size_t)r * dst_ld + c] = (__bf16)src[(size_t)r * src_ld + col0 + c];
    }
}

// per-column mean + rsqrt(var+eps) over `rows` (deterministic tree reduce)
__global__ void colstats_kernel(const float* __restrict__ src, int rows, int cols,
                                float* __restrict__ mu, float* __restrict__ rsig)
{
    int c = blockIdx.x;
    __shared__ float ss[256], ss2[256];
    float s = 0.f, s2 = 0.f;
    for (int r = threadIdx.x; r < rows; r += 256) {
        float v = src[(size_t)r * cols + c];
        s += v; s2 += v * v;
    }
    ss[threadIdx.x] = s; ss2[threadIdx.x] = s2;
    __syncthreads();
    for (int o = 128; o > 0; o >>= 1) {
        if (threadIdx.x < o) { ss[threadIdx.x] += ss[threadIdx.x + o];
                               ss2[threadIdx.x] += ss2[threadIdx.x + o]; }
        __syncthreads();
    }
    if (threadIdx.x == 0) {
        float m = ss[0] / rows;
        float var = ss2[0] / rows - m * m;
        mu[c] = m;
        rsig[c] = rsqrtf(var + 1e-5f);
    }
}

__global__ void bn_apply_kernel(const float* __restrict__ src,
                                const float* __restrict__ mu,
                                const float* __restrict__ rsig,
                                __bf16* __restrict__ dst, int rows, int cols)
{
    int idx = blockIdx.x * blockDim.x + threadIdx.x;
    if (idx < rows * cols) {
        int c = idx % cols;
        dst[idx] = (__bf16)((src[idx] - mu[c]) * rsig[c]);
    }
}

// final N=1 layer: per-row dot of length K
__global__ void lastdot_kernel(const __bf16* __restrict__ B2,
                               const float* __restrict__ w,
                               const float* __restrict__ b,
                               float* __restrict__ outv, int M, int K)
{
    int r = blockIdx.x * blockDim.x + threadIdx.x;
    if (r < M) {
        float s = b[0];
        for (int j = 0; j < K; ++j)
            s += (float)B2[(size_t)r * K + j] * w[j];
        outv[r] = s;
    }
}

// 32-pass MSB radix select of the (NN+1)-th largest value (single block)
__global__ void topk_thr_kernel(const float* __restrict__ vals, int n,
                                const int* __restrict__ NNp,
                                float* __restrict__ thr_out)
{
    __shared__ int s_cnt;
    __shared__ unsigned s_prefix;
    __shared__ int s_k;
    int tid = threadIdx.x;
    if (tid == 0) { s_prefix = 0u; s_k = NNp[0] + 1; }
    __syncthreads();
    for (int b = 31; b >= 0; --b) {
        if (tid == 0) s_cnt = 0;
        __syncthreads();
        unsigned want = (s_prefix | (1u << b)) >> b;
        int local = 0;
        for (int i = tid; i < n; i += blockDim.x) {
            unsigned u = __float_as_uint(vals[i]);
            unsigned t = (u & 0x80000000u) ? ~u : (u | 0x80000000u);
            if ((t >> b) == want) local++;
        }
        atomicAdd(&s_cnt, local);
        __syncthreads();
        if (tid == 0) {
            if (s_cnt >= s_k) s_prefix |= (1u << b);
            else              s_k -= s_cnt;
        }
        __syncthreads();
    }
    if (tid == 0) {
        unsigned t = s_prefix;
        unsigned u = (t & 0x80000000u) ? (t ^ 0x80000000u) : ~t;
        thr_out[0] = __uint_as_float(u);
    }
}

// out = v * where(v > thr, 1/v, 0)
__global__ void finalize_kernel(const float* __restrict__ mlp,
                                const float* __restrict__ thr,
                                float* __restrict__ out, int n)
{
    int i = blockIdx.x * blockDim.x + threadIdx.x;
    if (i < n) {
        float v = mlp[i];
        out[i] = (v > thr[0]) ? v * (1.0f / v) : 0.0f;
    }
}

// ---------------------------------------------------------------------------
extern "C" void kernel_launch(void* const* d_in, const int* in_sizes, int n_in,
                              void* d_out, int out_size, void* d_ws, size_t ws_size,
                              hipStream_t stream)
{
    (void)in_sizes; (void)n_in; (void)out_size; (void)ws_size;

    const int N = 10000, DT = 512, FT = 768;
    const int NHID = 256, NCLS = 128, NIN = 384, HH1 = 256, HH2 = 128;

    const float* x   = (const float*)d_in[0];
    const float* adj = (const float*)d_in[1];
    const float* gW1 = (const float*)d_in[2];
    const float* gb1 = (const float*)d_in[3];
    const float* gW2 = (const float*)d_in[4];
    const float* gb2 = (const float*)d_in[5];
    const float* gW3 = (const float*)d_in[6];
    const float* gb3 = (const float*)d_in[7];
    const float* lW1 = (const float*)d_in[8];
    const float* lb1 = (const float*)d_in[9];
    const float* lW2 = (const float*)d_in[10];
    const float* lb2 = (const float*)d_in[11];
    const float* lW3 = (const float*)d_in[12];
    const float* lb3 = (const float*)d_in[13];
    const int*   NNp = (const int*)d_in[15];

    // ---- workspace carve-up (256B aligned) ----
    char* ws = (char*)d_ws;
    size_t off = 0;
    auto alloc = [&](size_t bytes) {
        void* p = ws + off;
        off += (bytes + 255) & ~(size_t)255;
        return p;
    };
    __bf16* Xg   = (__bf16*)alloc((size_t)N * DT   * 2);
    __bf16* Tb   = (__bf16*)alloc((size_t)N * NHID * 2);
    __bf16* H1b  = (__bf16*)alloc((size_t)N * NHID * 2);
    __bf16* H2b  = (__bf16*)alloc((size_t)N * NHID * 2);
    __bf16* Cat  = (__bf16*)alloc((size_t)N * NIN  * 2);
    float*  M1   = (float*) alloc((size_t)N * HH1  * 4);
    __bf16* B1   = (__bf16*)alloc((size_t)N * HH1  * 2);
    float*  M2   = (float*) alloc((size_t)N * HH2  * 4);
    __bf16* B2   = (__bf16*)alloc((size_t)N * HH2  * 2);
    float*  mlp  = (float*) alloc((size_t)N * 4);
    __bf16* gW1b = (__bf16*)alloc((size_t)DT   * NHID * 2);
    __bf16* gW2b = (__bf16*)alloc((size_t)NHID * NHID * 2);
    __bf16* gW3b = (__bf16*)alloc((size_t)NHID * NCLS * 2);
    __bf16* lW1b = (__bf16*)alloc((size_t)NIN  * HH1  * 2);
    __bf16* lW2b = (__bf16*)alloc((size_t)HH1  * HH2  * 2);
    float*  mu1  = (float*) alloc(HH1 * 4);
    float*  rs1  = (float*) alloc(HH1 * 4);
    float*  mu2  = (float*) alloc(HH2 * 4);
    float*  rs2  = (float*) alloc(HH2 * 4);
    float*  thr  = (float*) alloc(4);

    auto cvt = [&](const float* s, int sld, int c0, __bf16* d, int dld, int r, int c) {
        int tot = r * c;
        cvt_submat_kernel<<<(tot + 255) / 256, 256, 0, stream>>>(s, sld, c0, d, dld, r, c);
    };

    // instantiations:
    //   adjG : A=f32 (adj), C=bf16, relu, bias
    //   wprj : A=bf16, C=bf16, no relu, no bias
    //   mlpG : A=bf16, C=f32, relu, bias
    auto adjG = [&](const float* A, const __bf16* B, const float* bias, __bf16* C,
                    int M_, int N_, int K_, int lda, int ldb, int ldc) {
        dim3 grid((N_ + BN - 1) / BN, (M_ + BM - 1) / BM);
        gemm_wmma_kernel<true, true, true><<<grid, 256, 0, stream>>>(
            A, B, bias, C, M_, N_, K_, lda, ldb, ldc);
    };
    auto wprj = [&](const __bf16* A, const __bf16* B, __bf16* C,
                    int M_, int N_, int K_, int lda, int ldb, int ldc) {
        dim3 grid((N_ + BN - 1) / BN, (M_ + BM - 1) / BM);
        gemm_wmma_kernel<false, true, false><<<grid, 256, 0, stream>>>(
            A, B, nullptr, C, M_, N_, K_, lda, ldb, ldc);
    };
    auto mlpG = [&](const __bf16* A, const __bf16* B, const float* bias, float* C,
                    int M_, int N_, int K_, int lda, int ldb, int ldc) {
        dim3 grid((N_ + BN - 1) / BN, (M_ + BM - 1) / BM);
        gemm_wmma_kernel<false, false, true><<<grid, 256, 0, stream>>>(
            A, B, bias, C, M_, N_, K_, lda, ldb, ldc);
    };

    // ---- conversions ----
    cvt(x, FT, 0,  Xg, DT, N, DT);                  // x[:, :512] -> bf16
    cvt(x, FT, DT, Cat + NCLS, NIN, N, FT - DT);    // x[:, 512:] -> concat cols 128..383
    cvt(gW1, NHID, 0, gW1b, NHID, DT,   NHID);
    cvt(gW2, NHID, 0, gW2b, NHID, NHID, NHID);
    cvt(gW3, NCLS, 0, gW3b, NCLS, NHID, NCLS);
    cvt(lW1, HH1,  0, lW1b, HH1,  NIN,  HH1);
    cvt(lW2, HH2,  0, lW2b, HH2,  HH1,  HH2);

    // ---- GCN layer 1 ----
    wprj(Xg, gW1b, Tb,        N, NHID, DT,   DT,   NHID, NHID);   // T1 = Xg@gW1
    adjG(adj, Tb, gb1, H1b,   N, NHID, N,    N,    NHID, NHID);   // H1 = relu(adj@T1+b)
    // ---- GCN layer 2 ----
    wprj(H1b, gW2b, Tb,       N, NHID, NHID, NHID, NHID, NHID);   // T2 = H1@gW2
    adjG(adj, Tb, gb2, H2b,   N, NHID, N,    N,    NHID, NHID);   // H2 = relu(adj@T2+b)
    // ---- GCN layer 3 (writes into concat cols 0..127) ----
    wprj(H2b, gW3b, Tb,       N, NCLS, NHID, NHID, NCLS, NCLS);   // T3 = H2@gW3
    adjG(adj, Tb, gb3, Cat,   N, NCLS, N,    N,    NCLS, NIN);    // H3 = relu(adj@T3+b)

    // ---- MLP layer 1 + BN ----
    mlpG(Cat, lW1b, lb1, M1,  N, HH1, NIN, NIN, HH1, HH1);
    colstats_kernel<<<HH1, 256, 0, stream>>>(M1, N, HH1, mu1, rs1);
    bn_apply_kernel<<<(N * HH1 + 255) / 256, 256, 0, stream>>>(M1, mu1, rs1, B1, N, HH1);
    // ---- MLP layer 2 + BN ----
    mlpG(B1, lW2b, lb2, M2,   N, HH2, HH1, HH1, HH2, HH2);
    colstats_kernel<<<HH2, 256, 0, stream>>>(M2, N, HH2, mu2, rs2);
    bn_apply_kernel<<<(N * HH2 + 255) / 256, 256, 0, stream>>>(M2, mu2, rs2, B2, N, HH2);
    // ---- final projection (N=1), top-k threshold, masking ----
    lastdot_kernel<<<(N + 255) / 256, 256, 0, stream>>>(B2, lW3, lb3, mlp, N, HH2);
    topk_thr_kernel<<<1, 1024, 0, stream>>>(mlp, N, NNp, thr);
    finalize_kernel<<<(N + 255) / 256, 256, 0, stream>>>(mlp, thr, (float*)d_out, N);
}
